// AutoRoundGPTQLinear_33337536152068
// MI455X (gfx1250) — compile-verified, matrix operand order checked by reading, not run on hardware
//
#include <hip/hip_runtime.h>

// ---------------- WMMA / vector types (CDNA5, wave32) ----------------
typedef __attribute__((ext_vector_type(16))) _Float16 v16h;
typedef __attribute__((ext_vector_type(8)))  _Float16 v8h;
typedef __attribute__((ext_vector_type(2)))  _Float16 h2;
typedef __attribute__((ext_vector_type(8)))  float    v8f;

#define IN_F   4096
#define OUT_F  4096
#define GROUP  128

#define BM     128
#define BN     128
#define BK     64
#define LDSK   (BK + 8)     // pad 8 halves => row stride 144B (16B-aligned, conflict-free)
#define NTHREADS 256        // 8 waves

union FragH {
    v16h v;
    v8h  h[2];
};

// pack two floats -> two f16 in one dword (v_cvt_pk_rtz_f16_f32)
static __device__ __forceinline__ unsigned f2h2(float a, float b) {
#if defined(__has_builtin) && __has_builtin(__builtin_amdgcn_cvt_pkrtz)
    auto r = __builtin_amdgcn_cvt_pkrtz(a, b);
    static_assert(sizeof(r) == 4, "unexpected cvt_pkrtz return size");
    return __builtin_bit_cast(unsigned, r);
#else
    h2 r = { (_Float16)a, (_Float16)b };
    return __builtin_bit_cast(unsigned, r);
#endif
}

static __device__ __forceinline__ unsigned byte_pair_perm(unsigned wo, unsigned we, int p) {
#if defined(__has_builtin) && __has_builtin(__builtin_amdgcn_perm)
    // result: byte0 = we.byte[p], byte2 = wo.byte[p], bytes 1/3 = 0   (one v_perm_b32)
    const unsigned sel = 0x0C000C00u | (unsigned)p | ((unsigned)(4 + p) << 16);
    return __builtin_amdgcn_perm(wo, we, sel);
#else
    return ((we >> (8 * p)) & 0xFFu) | (((wo >> (8 * p)) & 0xFFu) << 16);
#endif
}

__global__ __launch_bounds__(NTHREADS)
void gptq_f16_wmma_gemm(const float* __restrict__ x,
                        const int*   __restrict__ qw,
                        const int*   __restrict__ qz,
                        const _Float16* __restrict__ sc,
                        float* __restrict__ y) {
    __shared__ unsigned short As[2][BM * LDSK];   // x tile, f16, [m][k]
    __shared__ unsigned short Bs[2][BN * LDSK];   // dequant W tile, f16, [n][k]

    const int tid  = threadIdx.x;
    const int lane = tid & 31;
    const int wave = tid >> 5;          // 0..7
    const int waveM = wave & 3;         // 4 waves across M (32 rows each)
    const int waveN = wave >> 2;        // 2 waves across N (64 cols each)
    const int l16   = lane & 15;
    const int lhalf = lane >> 4;        // 0: lanes 0-15, 1: lanes 16-31

    const int n0 = blockIdx.x * BN;
    const int m0 = blockIdx.y * BM;

    // ---- A-tile loader mapping: 128x64 fp32; thread covers 8 consecutive k
    const int octA = tid & 7;           // column oct (8 floats)
    const int rowA = tid >> 3;          // 0..31, step 32

    // ---- B-tile loader mapping: 8 packed-k rows x 128 n -> 4 int32 per thread
    const int nB  = tid & 127;          // 0..127
    const int kq  = tid >> 7;           // 0 or 1 (handles k8 = kq*4..kq*4+3)
    const int gn  = n0 + nB;

    v8f acc[2][4];
#pragma unroll
    for (int i = 0; i < 2; ++i)
#pragma unroll
        for (int j = 0; j < 4; ++j)
            acc[i][j] = (v8f){0.f, 0.f, 0.f, 0.f, 0.f, 0.f, 0.f, 0.f};

    const unsigned aBase = (unsigned)((waveM * 32 + l16) * LDSK + lhalf * 8);
    const unsigned bBase = (unsigned)((waveN * 64 + l16) * LDSK + lhalf * 8);

    // staged tile held in registers between loadTile() and commitTile()
    float4   xa[4][2];
    unsigned wb[4];
    unsigned znib = 0;
    _Float16 sh = (_Float16)0.f;

    auto loadTile = [&](int kt) {
        const int k0 = kt * BK;
        const float* xp = x + (size_t)(m0 + rowA) * IN_F + k0 + octA * 8;
#pragma unroll
        for (int i = 0; i < 4; ++i) {
            xa[i][0] = *(const float4*)(xp + (size_t)(i * 32) * IN_F);      // global_load_b128
            xa[i][1] = *(const float4*)(xp + (size_t)(i * 32) * IN_F + 4);  // global_load_b128
        }
        const int g = k0 / GROUP;     // constant per K-tile (BK | GROUP)
        znib = (unsigned)((qz[g * (OUT_F / 8) + (gn >> 3)] >> (4 * (gn & 7))) & 15);
        sh   = sc[g * OUT_F + gn];
        const int* qwp = qw + (size_t)(k0 / 8 + kq * 4) * OUT_F + gn;
#pragma unroll
        for (int j = 0; j < 4; ++j)
            wb[j] = (unsigned)qwp[(size_t)j * OUT_F];
    };

    auto commitTile = [&](int buf) {
        // x: fp32 -> f16, LDS [m][k]
#pragma unroll
        for (int i = 0; i < 4; ++i) {
            uint4 d;
            d.x = f2h2(xa[i][0].x, xa[i][0].y);
            d.y = f2h2(xa[i][0].z, xa[i][0].w);
            d.z = f2h2(xa[i][1].x, xa[i][1].y);
            d.w = f2h2(xa[i][1].z, xa[i][1].w);
            *(uint4*)&As[buf][(rowA + i * 32) * LDSK + octA * 8] = d;   // ds_store_b128
        }
        // W: int4 -> f16 via 1024-bias trick; packed exact (h - (1024+z)) * s
        const h2 hz2 = __builtin_bit_cast(h2, 0x64006400u | znib | (znib << 16));
        const h2 s2  = { sh, sh };
#pragma unroll
        for (int j = 0; j < 4; ++j) {
            const unsigned we = wb[j] & 0x0F0F0F0Fu;          // even nibbles (k even)
            const unsigned wo = (wb[j] >> 4) & 0x0F0F0F0Fu;   // odd nibbles  (k odd)
            uint4 d;
            unsigned* dp = &d.x;
#pragma unroll
            for (int p = 0; p < 4; ++p) {
                const unsigned hbits = byte_pair_perm(wo, we, p) | 0x64006400u;
                const h2 hv = __builtin_bit_cast(h2, hbits);  // {1024+w_even, 1024+w_odd}
                const h2 r  = (hv - hz2) * s2;                // v_pk_add_f16(neg) + v_pk_mul_f16
                dp[p] = __builtin_bit_cast(unsigned, r);
            }
            *(uint4*)&Bs[buf][nB * LDSK + (kq * 4 + j) * 8] = d;   // ds_store_b128
        }
    };

    auto computeTile = [&](int buf) {
#pragma unroll
        for (int kk = 0; kk < BK; kk += 32) {
            FragH a[2], b[4];
#pragma unroll
            for (int mi = 0; mi < 2; ++mi) {
                const unsigned short* p = &As[buf][aBase + mi * 16 * LDSK + kk];
                a[mi].h[0] = *(const v8h*)p;             // ds_load_b128
                a[mi].h[1] = *(const v8h*)(p + 16);      // ds_load_b128
            }
#pragma unroll
            for (int nj = 0; nj < 4; ++nj) {
                const unsigned short* p = &Bs[buf][bBase + nj * 16 * LDSK + kk];
                b[nj].h[0] = *(const v8h*)p;
                b[nj].h[1] = *(const v8h*)(p + 16);
            }
#pragma unroll
            for (int mi = 0; mi < 2; ++mi)
#pragma unroll
                for (int nj = 0; nj < 4; ++nj)
                    acc[mi][nj] = __builtin_amdgcn_wmma_f32_16x16x32_f16(
                        false, a[mi].v, false, b[nj].v,
                        (short)0, acc[mi][nj], false, false);
        }
    };

    const int KT = IN_F / BK;           // 64 K-tiles
    // software pipeline: load(t+1) -> compute(t) -> commit(t+1) -> barrier
    loadTile(0);
    commitTile(0);
    __syncthreads();
    for (int kt = 0; kt < KT; ++kt) {
        const int buf = kt & 1;
        if (kt + 1 < KT) loadTile(kt + 1);      // HBM latency hides under WMMA
        computeTile(buf);
        if (kt + 1 < KT) commitTile(buf ^ 1);
        __syncthreads();
    }

    // -------- epilogue: C/D layout -> fp32 global --------
    // VGPR vg, lanes 0-15: (M=vg, N=l16); lanes 16-31: (M=vg+8, N=l16)
#pragma unroll
    for (int mi = 0; mi < 2; ++mi) {
#pragma unroll
        for (int nj = 0; nj < 4; ++nj) {
            const int col = n0 + waveN * 64 + nj * 16 + l16;
#pragma unroll
            for (int vg = 0; vg < 8; ++vg) {
                const int row = m0 + waveM * 32 + mi * 16 + lhalf * 8 + vg;
                y[(size_t)row * OUT_F + col] = acc[mi][nj][vg];
            }
        }
    }
}

extern "C" void kernel_launch(void* const* d_in, const int* in_sizes, int n_in,
                              void* d_out, int out_size, void* d_ws, size_t ws_size,
                              hipStream_t stream) {
    const float*    x  = (const float*)d_in[0];
    const int*      qw = (const int*)d_in[1];
    const int*      qz = (const int*)d_in[2];
    const _Float16* sc = (const _Float16*)d_in[3];
    float*          y  = (float*)d_out;

    const int Mtotal = in_sizes[0] / IN_F;    // 4*2048 = 8192

    dim3 grid(OUT_F / BN, Mtotal / BM);       // (32, 64)
    dim3 block(NTHREADS);
    gptq_f16_wmma_gemm<<<grid, block, 0, stream>>>(x, qw, qz, sc, y);

    (void)n_in; (void)out_size; (void)d_ws; (void)ws_size;
}
